// GTrXLBlock_66812511256963
// MI455X (gfx1250) — matrix operand order repaired and checked by reading, not compiled, not run on hardware
//
#include <hip/hip_runtime.h>
#include <hip/hip_bf16.h>

// ---------------------------------------------------------------------------
// GTrXL block for MI455X (gfx1250, wave32).
//  - All dense matmuls via v_wmma_f32_16x16x32_bf16 (f32 accumulate).
//  - Large-K GEMMs use an LDS-staged macro-tile (32M x 64N per 8-wave block)
//    with double-buffered GLOBAL_LOAD_ASYNC_TO_LDS_B128 (ASYNCcnt) when the
//    toolchain exposes the builtin; sync LDS staging otherwise.
//  - K=32 score GEMMs and N=32 AV GEMM use the direct per-wave path.
// ---------------------------------------------------------------------------

typedef __bf16 bf16_t;
typedef bf16_t v16bf __attribute__((ext_vector_type(16)));
typedef bf16_t v8bf  __attribute__((ext_vector_type(8)));
typedef float  v8f   __attribute__((ext_vector_type(8)));

#define QLEN   1024
#define MLEN   1024
#define KLEN   2048
#define DMODEL 256
#define NHEAD  8
#define DHEAD  32
#define FDIM   1024
#define G3     768   // 3*DMODEL gate width
#define APAD   72    // padded LDS row stride (elements): 144B = 36 banks

#if defined(__has_builtin)
#if __has_builtin(__builtin_amdgcn_global_load_async_to_lds_b128)
#define HAVE_ASYNC_LDS 1
#endif
#endif

#if defined(HAVE_ASYNC_LDS)
// Builtin signature (from compiler diagnostic): param1 is a GCC-style
// vector_size(16) int pointer in addrspace(1) (global src), param2 the LDS
// destination, then immediate offset + cpol.
typedef int v4i_vs __attribute__((vector_size(16)));
typedef __attribute__((address_space(1))) v4i_vs* async_gp_t;
typedef __attribute__((address_space(3))) v4i_vs* async_lp_t;
#endif

__device__ __forceinline__ void async_stage_wait() {
#if defined(HAVE_ASYNC_LDS)
#if __has_builtin(__builtin_amdgcn_s_wait_asynccnt)
  __builtin_amdgcn_s_wait_asynccnt(0);
#else
  asm volatile("s_wait_asynccnt 0x0" ::: "memory");
#endif
#endif
  __syncthreads();
}

// ---------------------------- elementwise helpers ---------------------------

__global__ void cvt_f32_bf16(const float* __restrict__ src,
                             bf16_t* __restrict__ dst, long n) {
  long i = (long)blockIdx.x * blockDim.x + threadIdx.x;
  if (i < n) dst[i] = (bf16_t)src[i];
}

__global__ void transpose_f32(const float* __restrict__ src,
                              float* __restrict__ dst, int rows, int cols) {
  long i = (long)blockIdx.x * blockDim.x + threadIdx.x;
  if (i < (long)rows * cols) {
    int r = (int)(i / cols), c = (int)(i % cols);
    dst[(long)c * rows + r] = src[i];
  }
}

// ----------------------- direct (per-wave) WMMA GEMM ------------------------
// C[z][m,n] = sum_k A[z][m,k]*B[z][n,k] (+bias[n]) (+resid[m,n]) (relu?)
// One wave per 16x16 C tile; fragments loaded straight from global (used for
// K=32 score GEMMs and the N=32 AV GEMM where staging has no reuse to win).
__global__ void __launch_bounds__(256)
gemm_bf16_wmma(const bf16_t* __restrict__ A, const bf16_t* __restrict__ B,
               const float* __restrict__ bias, const float* __restrict__ resid,
               float* __restrict__ C,
               int Mdim, int Ndim, int Kdim, int ldc, int ldResid,
               long strideA, long strideB, long strideC, int relu) {
  const int wave = threadIdx.x >> 5;
  const int lane = threadIdx.x & 31;
  const int tilesN = Ndim >> 4;
  const int tilesM = Mdim >> 4;
  const int tiles  = tilesM * tilesN;
  const int t = blockIdx.x * 8 + wave;
  if (t >= tiles) return;               // wave-uniform: EXEC stays all-1s
  const int tm = t / tilesN, tn = t % tilesN;
  const int z = blockIdx.y;

  const bf16_t* Ab = A + (long)z * strideA;
  const bf16_t* Bb = B + (long)z * strideB;
  float*        Cb = C + (long)z * strideC;

  const int r   = lane & 15;
  const int hiA = (lane >> 4) << 3;     // 0 or 8   (A K-group offset)
  const int hiB = (lane >> 4) << 4;     // 0 or 16  (unused: B loads use hiA)

  const bf16_t* arow = Ab + (long)(tm * 16 + r) * Kdim;
  const bf16_t* brow = Bb + (long)(tn * 16 + r) * Kdim;

  v8f acc = {};
  for (int k0 = 0; k0 < Kdim; k0 += 32) {
    v8bf a_lo = *(const v8bf*)(arow + k0 + hiA);
    v8bf a_hi = *(const v8bf*)(arow + k0 + 16 + hiA);
    v8bf b_lo = *(const v8bf*)(brow + k0 + hiA);
    v8bf b_hi = *(const v8bf*)(brow + k0 + 16 + hiA);
    v16bf avv, bvv;
#pragma unroll
    for (int e = 0; e < 8; ++e) {
      avv[e] = a_lo[e]; avv[e + 8] = a_hi[e];
      bvv[e] = b_lo[e]; bvv[e + 8] = b_hi[e];
    }
    acc = __builtin_amdgcn_wmma_f32_16x16x32_bf16(
        false, avv, false, bvv, (short)0, acc, false, false);
  }

  const int n = tn * 16 + r;
  const float bn = bias ? bias[n] : 0.0f;
  const int mBase = tm * 16 + ((lane >> 4) << 3);
#pragma unroll
  for (int e = 0; e < 8; ++e) {
    const int m = mBase + e;
    float vo = acc[e] + bn;
    if (resid) vo += resid[(long)m * ldResid + n];
    if (relu) vo = fmaxf(vo, 0.0f);
    Cb[(long)m * ldc + n] = vo;
  }
}

// ------------------- LDS-staged WMMA GEMM (macro 32M x 64N) -----------------
// Requires M%32==0, N%64==0, K%64==0. Block = 8 waves arranged 2(tm)x4(tn).
// K stepped in 64-wide double-buffered chunks staged to LDS with async copies.
__global__ void __launch_bounds__(256)
gemm_bf16_wmma_lds(const bf16_t* __restrict__ A, const bf16_t* __restrict__ B,
                   const float* __restrict__ bias, const float* __restrict__ resid,
                   float* __restrict__ C,
                   int Ndim, int Kdim, int ldc, int ldResid,
                   long strideA, long strideB, long strideC, int relu) {
  __shared__ bf16_t As[2][32 * APAD];   // 9.0 KB
  __shared__ bf16_t Bs[2][64 * APAD];   // 18.0 KB

  const int tid  = threadIdx.x;
  const int wave = tid >> 5;
  const int lane = tid & 31;
  const int macroN = Ndim >> 6;
  const int bm = blockIdx.x / macroN;
  const int bn = blockIdx.x % macroN;
  const int z  = blockIdx.y;

  const bf16_t* Ab = A + (long)z * strideA + (long)(bm * 32) * Kdim;
  const bf16_t* Bb = B + (long)z * strideB + (long)(bn * 64) * Kdim;
  float*        Cb = C + (long)z * strideC;

  // Stage 32 K columns (k0..k0+31): A 32x32 (128 x 16B) + B 64x32 (256 x 16B).
  auto stage_chunk = [&](int buf, int k0) {
    for (int idx = tid; idx < 384; idx += 256) {
      const int aSel = idx < 128;
      const int t2   = aSel ? idx : idx - 128;
      const int row  = t2 >> 2;
      const int ch   = (t2 & 3) * 8;       // element offset of 16B sub-chunk
      const bf16_t* g = (aSel ? Ab : Bb) + (long)row * Kdim + k0 + ch;
      bf16_t* l = aSel ? &As[buf][row * APAD + ch] : &Bs[buf][row * APAD + ch];
#if defined(HAVE_ASYNC_LDS)
      __builtin_amdgcn_global_load_async_to_lds_b128(
          (async_gp_t)g, (async_lp_t)l, 0, 0);
#else
      *(v8bf*)l = *(const v8bf*)g;
#endif
    }
  };

  const int r   = lane & 15;
  const int hiA = (lane >> 4) << 3;
  const int hiB = (lane >> 4) << 4;
  const int tmL = wave >> 2;              // 0..1
  const int tnL = wave & 3;               // 0..3

  v8f acc = {};
  const int nck = Kdim >> 6;
  stage_chunk(0, 0);
  async_stage_wait();
  for (int ck = 0; ck < nck; ++ck) {
    const int cur = ck & 1;
    if (ck + 1 < nck) stage_chunk(1 - cur, (ck + 1) << 6);
    const bf16_t* arow = &As[cur][(tmL * 16 + r) * APAD];
    const bf16_t* brow = &Bs[cur][(tnL * 16 + r) * APAD];
#pragma unroll
    for (int kk = 0; kk < 64; kk += 32) {
      v8bf a_lo = *(const v8bf*)(arow + kk + hiA);
      v8bf a_hi = *(const v8bf*)(arow + kk + 16 + hiA);
      v8bf b_lo = *(const v8bf*)(brow + kk + hiA);
      v8bf b_hi = *(const v8bf*)(brow + kk + 16 + hiA);
      v16bf avv, bvv;
#pragma unroll
      for (int e = 0; e < 8; ++e) {
        avv[e] = a_lo[e]; avv[e + 8] = a_hi[e];
        bvv[e] = b_lo[e]; bvv[e + 8] = b_hi[e];
      }
      acc = __builtin_amdgcn_wmma_f32_16x16x32_bf16(
          false, avv, false, bvv, (short)0, acc, false, false);
    }
    async_stage_wait();                   // next chunk landed; cur reusable
  }

  const int n = bn * 64 + tnL * 16 + r;
  const float bnv = bias ? bias[n] : 0.0f;
  const int mBase = bm * 32 + tmL * 16 + ((lane >> 4) << 3);
#pragma unroll
  for (int e = 0; e < 8; ++e) {
    const int m = mBase + e;
    float vo = acc[e] + bnv;
    if (resid) vo += resid[(long)m * ldResid + n];
    if (relu) vo = fmaxf(vo, 0.0f);
    Cb[(long)m * ldc + n] = vo;
  }
}

// --------------------------- head tensor builders ---------------------------

__global__ void build_q(const float* __restrict__ qkv, const float* __restrict__ u,
                        const float* __restrict__ v, bf16_t* __restrict__ qu,
                        bf16_t* __restrict__ qv) {
  int i = blockIdx.x;
  int hd = threadIdx.x;
  float q = qkv[(long)(MLEN + i) * G3 + hd];
  int h = hd >> 5, d = hd & 31;
  long o = (long)h * QLEN * DHEAD + (long)i * DHEAD + d;
  qu[o] = (bf16_t)(q + u[hd]);
  qv[o] = (bf16_t)(q + v[hd]);
}

__global__ void build_kv(const float* __restrict__ qkv,
                         bf16_t* __restrict__ kh, bf16_t* __restrict__ vT) {
  int j = blockIdx.x;
  int hd = threadIdx.x;
  int h = hd >> 5, d = hd & 31;
  float kk = qkv[(long)j * G3 + DMODEL + hd];
  float vv = qkv[(long)j * G3 + 2 * DMODEL + hd];
  kh[(long)h * KLEN * DHEAD + (long)j * DHEAD + d] = (bf16_t)kk;
  vT[(long)h * DHEAD * KLEN + (long)d * KLEN + j]  = (bf16_t)vv;
}

__global__ void build_rh(const float* __restrict__ rhf, bf16_t* __restrict__ rhb) {
  int j = blockIdx.x;
  int hd = threadIdx.x;
  int h = hd >> 5, d = hd & 31;
  rhb[(long)h * KLEN * DHEAD + (long)j * DHEAD + d] = (bf16_t)rhf[(long)j * DMODEL + hd];
}

// -------------- combine (AC + rel_shift(BD)), mask, softmax -----------------
__global__ void __launch_bounds__(256)
attn_softmax(const float* __restrict__ AC, const float* __restrict__ BD,
             float* __restrict__ attnF, bf16_t* __restrict__ attnB) {
  const int i = blockIdx.x, h = blockIdx.y, c = threadIdx.x;
  __shared__ float red[256];
  const long QKo = (long)h * QLEN * KLEN;
  float sc[8];
  float m = -3.0e38f;
#pragma unroll
  for (int s = 0; s < 8; ++s) {
    const int j = s * 256 + c;
    float ac = AC[QKo + (long)i * KLEN + j];
    // rel_shift: out[i,j] = xp_flat[q + i*k + j] on zero-padded (q, k+1)
    long tt  = (long)QLEN + (long)i * KLEN + j;
    long row = tt / (KLEN + 1);
    long col = tt % (KLEN + 1);
    float bd = (col == 0) ? 0.0f : BD[QKo + row * KLEN + (col - 1)];
    float sv = (ac + bd) * 0.17677669529663687f;   // 1/sqrt(32)
    if (j > i + MLEN) sv = -1.0e30f;               // causal-with-memory mask
    sc[s] = sv;
    m = fmaxf(m, sv);
  }
  red[c] = m; __syncthreads();
  for (int st = 128; st > 0; st >>= 1) {
    if (c < st) red[c] = fmaxf(red[c], red[c + st]);
    __syncthreads();
  }
  m = red[0]; __syncthreads();
  float ex[8], sum = 0.0f;
#pragma unroll
  for (int s = 0; s < 8; ++s) {
    ex[s] = (sc[s] <= -1.0e29f) ? 0.0f : __expf(sc[s] - m);
    sum += ex[s];
  }
  red[c] = sum; __syncthreads();
  for (int st = 128; st > 0; st >>= 1) {
    if (c < st) red[c] += red[c + st];
    __syncthreads();
  }
  const float inv = 1.0f / red[0];
#pragma unroll
  for (int s = 0; s < 8; ++s) {
    const int j = s * 256 + c;
    const float p = ex[s] * inv;
    attnF[((long)i * KLEN + j) * NHEAD + h] = p;     // (q,k,1,h) f32 output
    attnB[QKo + (long)i * KLEN + j] = (bf16_t)p;     // per-head bf16 for AV
  }
}

// ------------------------------ layernorm -----------------------------------
__global__ void __launch_bounds__(256)
layernorm(const float* __restrict__ x, const float* __restrict__ resid,
          const float* __restrict__ g, const float* __restrict__ b,
          float* __restrict__ outF, bf16_t* __restrict__ outB) {
  const int row = blockIdx.x, c = threadIdx.x;
  __shared__ float red[256];
  const float v = x[(long)row * DMODEL + c];
  red[c] = v; __syncthreads();
  for (int st = 128; st > 0; st >>= 1) {
    if (c < st) red[c] += red[c + st];
    __syncthreads();
  }
  const float mu = red[0] * (1.0f / DMODEL); __syncthreads();
  const float d = v - mu;
  red[c] = d * d; __syncthreads();
  for (int st = 128; st > 0; st >>= 1) {
    if (c < st) red[c] += red[c + st];
    __syncthreads();
  }
  const float var = red[0] * (1.0f / DMODEL);
  float y = d * rsqrtf(var + 1e-5f) * g[c] + b[c];
  if (resid) y += resid[(long)row * DMODEL + c];     // GRU input = resid + ln
  if (outF) outF[(long)row * DMODEL + c] = y;
  if (outB) outB[(long)row * DMODEL + c] = (bf16_t)y;
}

// ------------------------------ GRU scan ------------------------------------
// Single persistent workgroup (24 waves); sequential 1024-step recurrence.
// Whh is pre-transposed for coalesced reads (L2-resident); next-step gi row
// is prefetched (global_prefetch_b8) to hide the only streaming fetch.
__global__ void __launch_bounds__(768)
gru_scan(const float* __restrict__ gi, const float* __restrict__ WhhT,
         const float* __restrict__ bhh, float* __restrict__ ys) {
  const int tid = threadIdx.x;   // 0..767
  __shared__ float h[DMODEL];
  __shared__ float rg[DMODEL], zg[DMODEL], hn[DMODEL], inn[DMODEL];
  if (tid < DMODEL) h[tid] = 0.0f;
  __syncthreads();
  for (int t = 0; t < QLEN; ++t) {
    if (t + 1 < QLEN) __builtin_prefetch(&gi[(long)(t + 1) * G3 + tid], 0, 0);
    float acc = bhh[tid];
#pragma unroll 4
    for (int k = 0; k < DMODEL; ++k)
      acc = fmaf(WhhT[(long)k * G3 + tid], h[k], acc);   // coalesced across tid
    const float gv = gi[(long)t * G3 + tid];
    if (tid < 256)       rg[tid]       = 1.0f / (1.0f + __expf(-(gv + acc)));
    else if (tid < 512)  zg[tid - 256] = 1.0f / (1.0f + __expf(-(gv + acc)));
    else               { hn[tid - 512] = acc; inn[tid - 512] = gv; }
    __syncthreads();
    if (tid < 256) {
      const float ng = tanhf(inn[tid] + rg[tid] * hn[tid]);
      const float hv = (1.0f - zg[tid]) * ng + zg[tid] * h[tid];
      h[tid] = hv;
      ys[(long)t * DMODEL + tid] = hv;
    }
    __syncthreads();
  }
}

// ------------------------------ launcher ------------------------------------

extern "C" void kernel_launch(void* const* d_in, const int* in_sizes, int n_in,
                              void* d_out, int out_size, void* d_ws, size_t ws_size,
                              hipStream_t stream) {
  (void)in_sizes; (void)n_in; (void)out_size; (void)ws_size;

  const float* inputs = (const float*)d_in[0];
  const float* r      = (const float*)d_in[1];
  const float* u      = (const float*)d_in[2];
  const float* v      = (const float*)d_in[3];
  const float* mem    = (const float*)d_in[4];
  const float* ln1_g  = (const float*)d_in[5];
  const float* ln1_b  = (const float*)d_in[6];
  const float* ln2_g  = (const float*)d_in[7];
  const float* ln2_b  = (const float*)d_in[8];
  const float* Wqkv   = (const float*)d_in[9];
  const float* bqkv   = (const float*)d_in[10];
  const float* Wr     = (const float*)d_in[11];
  const float* Wo     = (const float*)d_in[12];
  const float* bo     = (const float*)d_in[13];
  const float* g1_Wih = (const float*)d_in[14];
  const float* g1_Whh = (const float*)d_in[15];
  const float* g1_bih = (const float*)d_in[16];
  const float* g1_bhh = (const float*)d_in[17];
  const float* g2_Wih = (const float*)d_in[18];
  const float* g2_Whh = (const float*)d_in[19];
  const float* g2_bih = (const float*)d_in[20];
  const float* g2_bhh = (const float*)d_in[21];
  const float* W1     = (const float*)d_in[22];
  const float* b1     = (const float*)d_in[23];
  const float* W2     = (const float*)d_in[24];
  const float* b2     = (const float*)d_in[25];
  // d_in[26] attn_mask: recomputed analytically (j > i + MLEN), not read.

  float* out_y    = (float*)d_out;                       // (1024,1,256)
  float* out_attn = out_y + (size_t)QLEN * DMODEL;       // (1024,2048,1,8)

  // ---- workspace layout ----
  char* wsb = (char*)d_ws;
  size_t off = 0;
  auto alloc = [&](size_t bytes) -> void* {
    void* p = wsb + off;
    off += (bytes + 255) & ~(size_t)255;
    return p;
  };
  bf16_t* cat_bf   = (bf16_t*)alloc((size_t)KLEN * DMODEL * 2);
  bf16_t* Wqkv_bf  = (bf16_t*)alloc((size_t)G3 * DMODEL * 2);
  bf16_t* r_bf     = (bf16_t*)alloc((size_t)KLEN * DMODEL * 2);
  bf16_t* Wr_bf    = (bf16_t*)alloc((size_t)DMODEL * DMODEL * 2);
  bf16_t* Wo_bf    = (bf16_t*)alloc((size_t)DMODEL * DMODEL * 2);
  bf16_t* W1_bf    = (bf16_t*)alloc((size_t)FDIM * DMODEL * 2);
  bf16_t* W2_bf    = (bf16_t*)alloc((size_t)DMODEL * FDIM * 2);
  bf16_t* g1Wih_bf = (bf16_t*)alloc((size_t)G3 * DMODEL * 2);
  bf16_t* g2Wih_bf = (bf16_t*)alloc((size_t)G3 * DMODEL * 2);
  float*  qkv_f    = (float*)alloc((size_t)KLEN * G3 * 4);
  float*  rh_f     = (float*)alloc((size_t)KLEN * DMODEL * 4);
  bf16_t* qu_bf    = (bf16_t*)alloc((size_t)NHEAD * QLEN * DHEAD * 2);
  bf16_t* qv_bf    = (bf16_t*)alloc((size_t)NHEAD * QLEN * DHEAD * 2);
  bf16_t* kh_bf    = (bf16_t*)alloc((size_t)NHEAD * KLEN * DHEAD * 2);
  bf16_t* vT_bf    = (bf16_t*)alloc((size_t)NHEAD * DHEAD * KLEN * 2);
  bf16_t* rh_bf    = (bf16_t*)alloc((size_t)NHEAD * KLEN * DHEAD * 2);
  float*  AC_f     = (float*)alloc((size_t)NHEAD * QLEN * KLEN * 4);
  float*  BD_f     = (float*)alloc((size_t)NHEAD * QLEN * KLEN * 4);
  bf16_t* attn_bf  = (bf16_t*)alloc((size_t)NHEAD * QLEN * KLEN * 2);
  float*  av_f     = (float*)alloc((size_t)QLEN * DMODEL * 4);
  bf16_t* av_bf    = (bf16_t*)alloc((size_t)QLEN * DMODEL * 2);
  float*  y_f      = (float*)alloc((size_t)QLEN * DMODEL * 4);
  bf16_t* xg1_bf   = (bf16_t*)alloc((size_t)QLEN * DMODEL * 2);
  float*  gi_f     = (float*)alloc((size_t)QLEN * G3 * 4);
  float*  x2_f     = (float*)alloc((size_t)QLEN * DMODEL * 4);
  bf16_t* xm_bf    = (bf16_t*)alloc((size_t)QLEN * DMODEL * 2);
  float*  h1_f     = (float*)alloc((size_t)QLEN * FDIM * 4);
  bf16_t* h1_bf    = (bf16_t*)alloc((size_t)QLEN * FDIM * 2);
  float*  xg2_f    = (float*)alloc((size_t)QLEN * DMODEL * 4);
  bf16_t* xg2_bf   = (bf16_t*)alloc((size_t)QLEN * DMODEL * 2);
  float*  WhhT1    = (float*)alloc((size_t)G3 * DMODEL * 4);
  float*  WhhT2    = (float*)alloc((size_t)G3 * DMODEL * 4);

  auto cvt = [&](const float* s, bf16_t* d, long n) {
    cvt_f32_bf16<<<dim3((unsigned)((n + 255) / 256)), 256, 0, stream>>>(s, d, n);
  };
  // direct path: K=32 score GEMMs, N=32 AV GEMM
  auto gemmD = [&](const bf16_t* A, const bf16_t* B, const float* bias,
                   const float* resid, float* C, int M, int N, int K,
                   int ldc, int ldResid, long sA, long sB, long sC,
                   int batch, int relu) {
    int tiles = (M / 16) * (N / 16);
    dim3 g((unsigned)((tiles + 7) / 8), (unsigned)batch);
    gemm_bf16_wmma<<<g, 256, 0, stream>>>(A, B, bias, resid, C, M, N, K, ldc,
                                          ldResid, sA, sB, sC, relu);
  };
  // LDS-staged path: M%32==0, N%64==0, K%64==0
  auto gemmS = [&](const bf16_t* A, const bf16_t* B, const float* bias,
                   const float* resid, float* C, int M, int N, int K,
                   int ldc, int ldResid, int relu) {
    dim3 g((unsigned)((M / 32) * (N / 64)), 1);
    gemm_bf16_wmma_lds<<<g, 256, 0, stream>>>(A, B, bias, resid, C, N, K, ldc,
                                              ldResid, 0, 0, 0, relu);
  };

  // ---- precision conversion + weight prep ----
  cvt(mem,    cat_bf,                         (long)MLEN * DMODEL);
  cvt(inputs, cat_bf + (size_t)MLEN * DMODEL, (long)QLEN * DMODEL);
  cvt(r,      r_bf,    (long)KLEN * DMODEL);
  cvt(Wqkv,   Wqkv_bf, (long)G3 * DMODEL);
  cvt(Wr,     Wr_bf,   (long)DMODEL * DMODEL);
  cvt(Wo,     Wo_bf,   (long)DMODEL * DMODEL);
  cvt(W1,     W1_bf,   (long)FDIM * DMODEL);
  cvt(W2,     W2_bf,   (long)DMODEL * FDIM);
  cvt(g1_Wih, g1Wih_bf, (long)G3 * DMODEL);
  cvt(g2_Wih, g2Wih_bf, (long)G3 * DMODEL);
  {
    unsigned blocks = (unsigned)(((long)G3 * DMODEL + 255) / 256);
    transpose_f32<<<blocks, 256, 0, stream>>>(g1_Whh, WhhT1, G3, DMODEL);
    transpose_f32<<<blocks, 256, 0, stream>>>(g2_Whh, WhhT2, G3, DMODEL);
  }

  // ---- attention ----
  gemmS(cat_bf, Wqkv_bf, bqkv, nullptr, qkv_f, KLEN, G3, DMODEL, G3, 0, 0);
  gemmS(r_bf, Wr_bf, nullptr, nullptr, rh_f, KLEN, DMODEL, DMODEL, DMODEL, 0, 0);
  build_q <<<dim3(QLEN), 256, 0, stream>>>(qkv_f, u, v, qu_bf, qv_bf);
  build_kv<<<dim3(KLEN), 256, 0, stream>>>(qkv_f, kh_bf, vT_bf);
  build_rh<<<dim3(KLEN), 256, 0, stream>>>(rh_f, rh_bf);
  // AC[h] = (q+u)_h @ k_h^T ; BD[h] = (q+v)_h @ rh_h^T   (K=32 -> direct)
  gemmD(qu_bf, kh_bf, nullptr, nullptr, AC_f, QLEN, KLEN, DHEAD, KLEN, 0,
        (long)QLEN * DHEAD, (long)KLEN * DHEAD, (long)QLEN * KLEN, NHEAD, 0);
  gemmD(qv_bf, rh_bf, nullptr, nullptr, BD_f, QLEN, KLEN, DHEAD, KLEN, 0,
        (long)QLEN * DHEAD, (long)KLEN * DHEAD, (long)QLEN * KLEN, NHEAD, 0);
  attn_softmax<<<dim3(QLEN, NHEAD), 256, 0, stream>>>(AC_f, BD_f, out_attn,
                                                      attn_bf);
  // av[h] = attn_h (1024x2048) @ v_h via vT (32x2048)   (N=32 -> direct)
  gemmD(attn_bf, vT_bf, nullptr, nullptr, av_f, QLEN, DHEAD, KLEN, DMODEL, 0,
        (long)QLEN * KLEN, (long)DHEAD * KLEN, (long)DHEAD, NHEAD, 0);
  cvt(av_f, av_bf, (long)QLEN * DMODEL);
  gemmS(av_bf, Wo_bf, bo, nullptr, y_f, QLEN, DMODEL, DMODEL, DMODEL, 0, 0);

  // ---- gated block: LN1 -> GRU1 -> LN2 -> MLP -> GRU2 ----
  layernorm<<<dim3(QLEN), 256, 0, stream>>>(y_f, inputs, ln1_g, ln1_b,
                                            nullptr, xg1_bf);
  gemmS(xg1_bf, g1Wih_bf, g1_bih, nullptr, gi_f, QLEN, G3, DMODEL, G3, 0, 0);
  gru_scan<<<dim3(1), 768, 0, stream>>>(gi_f, WhhT1, g1_bhh, x2_f);

  layernorm<<<dim3(QLEN), 256, 0, stream>>>(x2_f, nullptr, ln2_g, ln2_b,
                                            nullptr, xm_bf);
  gemmS(xm_bf, W1_bf, b1, nullptr, h1_f, QLEN, FDIM, DMODEL, FDIM, 0, 1);
  cvt(h1_f, h1_bf, (long)QLEN * FDIM);
  gemmS(h1_bf, W2_bf, b2, x2_f, xg2_f, QLEN, DMODEL, FDIM, DMODEL, DMODEL, 0);
  cvt(xg2_f, xg2_bf, (long)QLEN * DMODEL);
  gemmS(xg2_bf, g2Wih_bf, g2_bih, nullptr, gi_f, QLEN, G3, DMODEL, G3, 0, 0);
  gru_scan<<<dim3(1), 768, 0, stream>>>(gi_f, WhhT2, g2_bhh, out_y);
}
